// LinearAttentionTransformer_33517924778702
// MI455X (gfx1250) — compile-verified, matrix operand order checked by reading, not run on hardware
//
#include <hip/hip_runtime.h>
#include <hip/hip_bf16.h>

#ifndef __has_builtin
#define __has_builtin(x) 0
#endif

#define DEV __device__ __forceinline__

// ---------------- problem dimensions (compile-time) ----------------
constexpr int Bsz   = 2;
constexpr int Nseq  = 4096;
constexpr int DIMC  = 1024;
constexpr int DEPTH = 6;
constexpr int HEADS = 8;
constexpr int DHEAD = 128;
constexpr int BUCKET= 64;
constexpr int DFF   = 4096;
constexpr int HD    = HEADS * DHEAD;       // 1024
constexpr int NT    = Bsz * Nseq;          // 8192 tokens
constexpr int U     = Nseq / BUCKET;       // 64 buckets
constexpr int BH    = Bsz * HEADS;         // 16

typedef __bf16 bf16;
typedef __attribute__((ext_vector_type(16))) __bf16 v16bf;
typedef __attribute__((ext_vector_type(8)))  float  v8f;
typedef __attribute__((ext_vector_type(4)))  unsigned int v4u;
typedef __attribute__((ext_vector_type(8)))  int    v8i;
typedef __attribute__((ext_vector_type(4)))  int    v4i;

// ---------------- TDM (Tensor Data Mover) tile load ----------------
#if __has_builtin(__builtin_amdgcn_tensor_load_to_lds)
#define USE_TDM 1
#else
#define USE_TDM 0
#endif

DEV void wait_tensor0() {
#if __has_builtin(__builtin_amdgcn_s_wait_tensorcnt)
  __builtin_amdgcn_s_wait_tensorcnt(0);
#else
  asm volatile("s_wait_tensorcnt 0x0" ::: "memory");
#endif
}

#if USE_TDM
// Load a [rows x 64] bf16 tile (row stride = ld elements) from global into
// contiguous LDS at lds_addr. 2D D#: group0 = {count/lds/global/type},
// group1 = {data_size=2B, tensor dims, tile dims, dim0 stride}.
DEV void tdm_load_tile(unsigned lds_addr, const bf16* g, int ld, int rows) {
  unsigned long long ga = (unsigned long long)(size_t)g;
  v4u g0;
  g0[0] = 1u;                                    // count=1, user descriptor
  g0[1] = lds_addr;                              // lds_addr  [63:32]
  g0[2] = (unsigned)(ga & 0xFFFFFFFFu);          // global_addr lo
  g0[3] = (unsigned)((ga >> 32) & 0x1FFFFFFu)    // global_addr hi (57-bit)
        | (2u << 30);                            // type = 2 ("image")
  unsigned tdim0 = (unsigned)ld;                 // tensor width (elems)
  unsigned tdim1 = 1u << 20;                     // tall enough; tiles in-bounds
  v8i g1;
  g1[0] = (int)(1u << 16);                       // data_size = 1 (2 bytes)
  g1[1] = (int)((tdim0 & 0xFFFFu) << 16);        // tensor_dim0 [15:0]
  g1[2] = (int)((tdim0 >> 16) | ((tdim1 & 0xFFFFu) << 16));
  g1[3] = (int)((tdim1 >> 16) | (64u << 16));    // tile_dim0 = 64 elems
  g1[4] = (int)((unsigned)rows);                 // tile_dim1 = rows, tile_dim2=0
  g1[5] = (int)((unsigned)ld);                   // tensor_dim0_stride lo32
  g1[6] = 0;
  g1[7] = 0;
  v4i z4 = {0, 0, 0, 0};
#if __clang_major__ >= 23
  v8i z8 = {0, 0, 0, 0, 0, 0, 0, 0};
  __builtin_amdgcn_tensor_load_to_lds(g0, g1, z4, z4, z8, 0);
#else
  __builtin_amdgcn_tensor_load_to_lds(g0, g1, z4, z4, 0);
#endif
}
#endif

// Cooperative fallback loader (global -> LDS), 16B per thread-iteration.
template<int ROWS>
DEV void coop_load64(bf16* dst, const bf16* src, int ld, int tid) {
  constexpr int CH = ROWS * 8;                   // 8 x 16B chunks per 64-elem row
  for (int i = tid; i < CH; i += 256) {
    int r = i >> 3, c = (i & 7) << 3;
    *(uint4*)(dst + r * 64 + c) = *(const uint4*)(src + (size_t)r * ld + c);
  }
}

// ---------------- WMMA fragment helpers ----------------
union Frag16 { v16bf v; uint4 q[2]; };

// A/B fragment from LDS tile laid out [rows][64] bf16 (B staged transposed, so
// both operands use the 16-bit A-matrix lane pattern: lane<16 -> K 0..7/16..23,
// lane>=16 -> K 8..15/24..31, all relative to k-step base kk).
DEV v16bf load_frag64(const bf16* smem, int rowbase, int kk, int lane) {
  int r  = rowbase + (lane & 15);
  int kb = kk + ((lane >> 4) << 3);
  Frag16 f;
  f.q[0] = *(const uint4*)(smem + r * 64 + kb);
  f.q[1] = *(const uint4*)(smem + r * 64 + kb + 16);
  return f.v;
}

// ---------------- epilogue modes ----------------
enum {
  M_ST = 0,        // raw f32, batched [z][BM][BN]   (attention phase 1)
  M_BIASRES,       // f32: acc + bias[col] + resid   (Wo proj, FFN-2)
  M_GELU,          // bf16: gelu(acc + bias[col])    (FFN-1)
  M_Q,             // bf16 -> q [b,h,n,e]
  M_KEXPT,         // bf16 -> exp(k) transposed [b,h,e,n]
  M_VT,            // bf16 -> v transposed [b,h,e,n]
  M_AOUT           // bf16: acc * dinv[row] -> a [b,n,h*e]  (attention phase 3)
};

// ---------------- generic batched WMMA GEMM ----------------
// C[M,N] = A[M,K] (bf16 row-major, lda) x B[K,N] with B staged transposed:
// Bt[N,K] (bf16 row-major, ldb).  blockIdx.z batches: ptr += zh*z?1 + zu*z?2.
// Double-buffered LDS: TDM loads slab t+1 while WMMA consumes slab t; the only
// per-iteration sync is one s_wait_tensorcnt (wave 0) + one split barrier.
template<int BM, int BN, int WM, int WN, int MODE>
__global__ __launch_bounds__(256) void wmma_gemm(
    const bf16* __restrict__ A, int lda, long long zA1, long long zA2,
    const bf16* __restrict__ Bt, int ldb, long long zB1, long long zB2,
    int zdiv, int K,
    float* __restrict__ outF, bf16* __restrict__ outH,
    const float* __restrict__ bias, const float* __restrict__ resid,
    const float* __restrict__ dinv) {
  constexpr int BK  = 64;                 // two WMMA k-steps per slab
  constexpr int WTM = BM / WM;
  constexpr int WTN = BN / WN;
  constexpr int TM  = WTM / 16;
  constexpr int TN  = WTN / 16;

  __shared__ bf16 sA[2][BM][BK];
  __shared__ bf16 sB[2][BN][BK];

  const int tid  = threadIdx.x;
  const int wave = tid >> 5;
  const int lane = tid & 31;
  const int wm   = wave / WN;
  const int wn   = wave % WN;

  const int z  = blockIdx.z;
  const int zh = z / zdiv;
  const int zu = z % zdiv;
  const int m0 = blockIdx.y * BM;
  const int n0 = blockIdx.x * BN;
  const bf16* Aptr = A  + (size_t)zh * zA1 + (size_t)zu * zA2 + (size_t)m0 * lda;
  const bf16* Bptr = Bt + (size_t)zh * zB1 + (size_t)zu * zB2 + (size_t)n0 * ldb;

  v8f acc[TM][TN] = {};
  const int nk = K / BK;

#if USE_TDM
  if (wave == 0) {                         // prologue: slab 0 in flight
    tdm_load_tile((unsigned)(size_t)&sA[0][0][0], Aptr, lda, BM);
    tdm_load_tile((unsigned)(size_t)&sB[0][0][0], Bptr, ldb, BN);
  }
#endif
  for (int t = 0; t < nk; ++t) {
    const int cur = t & 1;
#if USE_TDM
    if (wave == 0) wait_tensor0();         // slab t landed in LDS
    __syncthreads();                       // publish slab t; slab t-1 free
    if (wave == 0 && t + 1 < nk) {         // overlap: DMA slab t+1 vs compute t
      tdm_load_tile((unsigned)(size_t)&sA[cur ^ 1][0][0],
                    Aptr + (size_t)(t + 1) * BK, lda, BM);
      tdm_load_tile((unsigned)(size_t)&sB[cur ^ 1][0][0],
                    Bptr + (size_t)(t + 1) * BK, ldb, BN);
    }
#else
    __syncthreads();
    coop_load64<BM>(&sA[cur][0][0], Aptr + (size_t)t * BK, lda, tid);
    coop_load64<BN>(&sB[cur][0][0], Bptr + (size_t)t * BK, ldb, tid);
    __syncthreads();
#endif
#pragma unroll
    for (int kk = 0; kk < BK; kk += 32) {
      v16bf af[TM], bfg[TN];
#pragma unroll
      for (int i = 0; i < TM; i++)
        af[i]  = load_frag64(&sA[cur][0][0], wm * WTM + i * 16, kk, lane);
#pragma unroll
      for (int j = 0; j < TN; j++)
        bfg[j] = load_frag64(&sB[cur][0][0], wn * WTN + j * 16, kk, lane);
#pragma unroll
      for (int i = 0; i < TM; i++)
#pragma unroll
        for (int j = 0; j < TN; j++)
          acc[i][j] = __builtin_amdgcn_wmma_f32_16x16x32_bf16(
              false, af[i], false, bfg[j], (short)0, acc[i][j], false, false);
    }
  }

  // ---------------- epilogue ----------------
#pragma unroll
  for (int i = 0; i < TM; i++) {
#pragma unroll
    for (int j = 0; j < TN; j++) {
      const int rb = m0 + wm * WTM + i * 16 + ((lane >> 4) << 3);
      const int cb = n0 + wn * WTN + j * 16 + (lane & 15);
#pragma unroll
      for (int e8 = 0; e8 < 8; e8++) {
        const int row = rb + e8;
        const int col = cb;
        const float val = acc[i][j][e8];
        if constexpr (MODE == M_ST) {
          outF[(size_t)z * (BM * BN) + (size_t)row * BN + col] = val;
        } else if constexpr (MODE == M_BIASRES) {
          const size_t o = (size_t)row * DIMC + col;
          outF[o] = val + bias[col] + resid[o];
        } else if constexpr (MODE == M_GELU) {
          const float t = val + bias[col];
          outH[(size_t)row * DFF + col] =
              (bf16)(0.5f * t * (1.f + erff(t * 0.70710678118654752f)));
        } else if constexpr (MODE == M_Q) {
          const int b = row >> 12, n = row & (Nseq - 1);
          const int h = col >> 7,  e = col & 127;
          outH[(((size_t)(b * HEADS + h) * Nseq + n) << 7) + e] = (bf16)val;
        } else if constexpr (MODE == M_KEXPT) {
          const int b = row >> 12, n = row & (Nseq - 1);
          const int h = col >> 7,  e = col & 127;
          outH[((size_t)(b * HEADS + h) * DHEAD + e) * Nseq + n] = (bf16)expf(val);
        } else if constexpr (MODE == M_VT) {
          const int b = row >> 12, n = row & (Nseq - 1);
          const int h = col >> 7,  e = col & 127;
          outH[((size_t)(b * HEADS + h) * DHEAD + e) * Nseq + n] = (bf16)val;
        } else if constexpr (MODE == M_AOUT) {
          const int b = zh / HEADS, hh = zh % HEADS;
          const int n = zu * BUCKET + row;
          const int tok = b * Nseq + n;
          const float sc = dinv[(size_t)zh * Nseq + n];
          outH[(size_t)tok * HD + hh * DHEAD + col] = (bf16)(val * sc);
        }
      }
    }
  }
}

// ---------------- weight cast+transpose: f32 [K,N] -> bf16 [N,K] ----------------
__global__ __launch_bounds__(256) void castT_kernel(const float* __restrict__ in,
                                                    bf16* __restrict__ out,
                                                    int K, int N) {
  __shared__ float tile[32][33];
  const int n0 = blockIdx.x * 32, k0 = blockIdx.y * 32;
  for (int i = threadIdx.y; i < 32; i += 8)
    tile[i][threadIdx.x] = in[(size_t)(k0 + i) * N + n0 + threadIdx.x];
  __syncthreads();
  for (int i = threadIdx.y; i < 32; i += 8)
    out[(size_t)(n0 + i) * K + k0 + threadIdx.x] = (bf16)tile[threadIdx.x][i];
}

// ---------------- layernorm: f32 row -> bf16 row ----------------
__global__ __launch_bounds__(256) void layernorm_kernel(const float* __restrict__ x,
                                                        const float* __restrict__ g,
                                                        const float* __restrict__ b,
                                                        bf16* __restrict__ out) {
  const int t = blockIdx.x;
  const float* row = x + (size_t)t * DIMC;
  float v[4], s = 0.f, s2 = 0.f;
#pragma unroll
  for (int i = 0; i < 4; i++) {
    v[i] = row[threadIdx.x + 256 * i];
    s += v[i]; s2 += v[i] * v[i];
  }
  for (int off = 16; off; off >>= 1) { s += __shfl_xor(s, off); s2 += __shfl_xor(s2, off); }
  __shared__ float rs[8], rs2[8];
  if ((threadIdx.x & 31) == 0) { rs[threadIdx.x >> 5] = s; rs2[threadIdx.x >> 5] = s2; }
  __syncthreads();
  s = 0.f; s2 = 0.f;
#pragma unroll
  for (int i = 0; i < 8; i++) { s += rs[i]; s2 += rs2[i]; }
  const float mean = s * (1.f / DIMC);
  const float var  = s2 * (1.f / DIMC) - mean * mean;
  const float rstd = rsqrtf(var + 1e-5f);
  bf16* o = out + (size_t)t * DIMC;
#pragma unroll
  for (int i = 0; i < 4; i++) {
    const int c = threadIdx.x + 256 * i;
    o[c] = (bf16)((v[i] - mean) * rstd * g[c] + b[c]);
  }
}

// ---------------- q: softmax over feature dim * e^-0.5 (in place) ----------------
__global__ __launch_bounds__(256) void softmax_q_kernel(bf16* __restrict__ q) {
  const int row  = blockIdx.x * 8 + (threadIdx.x >> 5);
  const int lane = threadIdx.x & 31;
  bf16* p = q + (size_t)row * DHEAD;
  float v[4];
#pragma unroll
  for (int i = 0; i < 4; i++) v[i] = (float)p[lane + 32 * i];
  float m = fmaxf(fmaxf(v[0], v[1]), fmaxf(v[2], v[3]));
  for (int off = 16; off; off >>= 1) m = fmaxf(m, __shfl_xor(m, off));
  float s = 0.f;
#pragma unroll
  for (int i = 0; i < 4; i++) { v[i] = expf(v[i] - m); s += v[i]; }
  for (int off = 16; off; off >>= 1) s += __shfl_xor(s, off);
  const float sc = 0.08838834764831845f / s;   // 128^-0.5 / sum
#pragma unroll
  for (int i = 0; i < 4; i++) p[lane + 32 * i] = (bf16)(v[i] * sc);
}

// ---------------- per-bucket k sums: ks[bh,u,d] = sum_n exp(k)[bh,d,u*64+n] ----------------
__global__ __launch_bounds__(256) void ksum_kernel(const bf16* __restrict__ kT,
                                                   float* __restrict__ ks) {
  const int id = blockIdx.x * 256 + threadIdx.x;     // BH*U*DHEAD
  const int d  = id % DHEAD;
  const int u  = (id / DHEAD) % U;
  const int bh = id / (DHEAD * U);
  const bf16* p = kT + ((size_t)bh * DHEAD + d) * Nseq + u * BUCKET;
  float s = 0.f;
#pragma unroll 4
  for (int i = 0; i < BUCKET; i++) s += (float)p[i];
  ks[id] = s;
}

// ---------------- exclusive cumsum over buckets (blindspot shift) ----------------
__global__ __launch_bounds__(256) void cumsum_ctx_kernel(const float* __restrict__ ST,
                                                         bf16* __restrict__ ctx) {
  const int bh   = blockIdx.x >> 6;
  const int slot = ((blockIdx.x & 63) << 8) + threadIdx.x;   // 0..16383
  const size_t base = (size_t)bh * U * 16384 + slot;
  float run = 0.f;
  for (int u = 0; u < U; u++) {
    const size_t idx = base + (size_t)u * 16384;
    ctx[idx] = (bf16)run;
    run += ST[idx];
  }
}

__global__ __launch_bounds__(256) void cumsum_ks_kernel(const float* __restrict__ ks,
                                                        float* __restrict__ kcum) {
  const int id = blockIdx.x * 256 + threadIdx.x;   // BH*DHEAD
  const int bh = id >> 7, d = id & 127;
  const size_t base = (size_t)bh * U * DHEAD + d;
  float run = 0.f;
  for (int u = 0; u < U; u++) {
    const size_t idx = base + (size_t)u * DHEAD;
    kcum[idx] = run;
    run += ks[idx];
  }
}

// ---------------- D^-1 = 1/max(q . kcum, eps) ----------------
__global__ __launch_bounds__(256) void dinv_kernel(const bf16* __restrict__ q,
                                                   const float* __restrict__ kcum,
                                                   float* __restrict__ dv) {
  const int id = blockIdx.x * 256 + threadIdx.x;   // BH*Nseq
  const int bh = id / Nseq, n = id % Nseq, u = n / BUCKET;
  const bf16*  qp = q + (size_t)id * DHEAD;
  const float* kc = kcum + ((size_t)bh * U + u) * DHEAD;
  float s = 0.f;
#pragma unroll 8
  for (int d = 0; d < DHEAD; d++) s += (float)qp[d] * kc[d];
  dv[id] = 1.f / fmaxf(s, 1e-3f);
}

// ---------------- host orchestration ----------------
extern "C" void kernel_launch(void* const* d_in, const int* in_sizes, int n_in,
                              void* d_out, int out_size, void* d_ws, size_t ws_size,
                              hipStream_t stream) {
  (void)in_sizes; (void)n_in; (void)out_size; (void)ws_size;
  const float* x_in = (const float*)d_in[0];
  const float* ln1g = (const float*)d_in[1];
  const float* ln1b = (const float*)d_in[2];
  const float* Wq   = (const float*)d_in[3];
  const float* Wk   = (const float*)d_in[4];
  const float* Wv   = (const float*)d_in[5];
  const float* Wo   = (const float*)d_in[6];
  const float* bo   = (const float*)d_in[7];
  const float* ln2g = (const float*)d_in[8];
  const float* ln2b = (const float*)d_in[9];
  const float* W1   = (const float*)d_in[10];
  const float* b1   = (const float*)d_in[11];
  const float* W2   = (const float*)d_in[12];
  const float* b2   = (const float*)d_in[13];

  float* xbuf = (float*)d_out;                // running residual stream (= output)

  char* ws = (char*)d_ws;
  size_t off = 0;
  auto alloc = [&](size_t bytes) -> void* {
    void* p = ws + off; off += (bytes + 255) & ~(size_t)255; return p;
  };
  bf16*  wTq  = (bf16*) alloc((size_t)HD  * DIMC * 2);   // W^T per-layer scratch
  bf16*  wTk  = (bf16*) alloc((size_t)HD  * DIMC * 2);
  bf16*  wTv  = (bf16*) alloc((size_t)HD  * DIMC * 2);
  bf16*  wTo  = (bf16*) alloc((size_t)DIMC* HD   * 2);
  bf16*  wT1  = (bf16*) alloc((size_t)DFF * DIMC * 2);
  bf16*  wT2  = (bf16*) alloc((size_t)DIMC* DFF  * 2);
  bf16*  hbuf = (bf16*) alloc((size_t)NT * DIMC * 2);    // LN output
  bf16*  qbuf = (bf16*) alloc((size_t)BH * Nseq * DHEAD * 2);
  bf16*  kT   = (bf16*) alloc((size_t)BH * DHEAD * Nseq * 2);
  bf16*  vT   = (bf16*) alloc((size_t)BH * DHEAD * Nseq * 2);
  bf16*  abuf = (bf16*) alloc((size_t)NT * HD * 2);
  bf16*  y1   = (bf16*) alloc((size_t)NT * DFF * 2);
  float* STb  = (float*)alloc((size_t)BH * U * 16384 * 4);
  bf16*  ctxb = (bf16*) alloc((size_t)BH * U * 16384 * 2);
  float* ksb  = (float*)alloc((size_t)BH * U * DHEAD * 4);
  float* kcum = (float*)alloc((size_t)BH * U * DHEAD * 4);
  float* dvb  = (float*)alloc((size_t)BH * Nseq * 4);

  hipMemcpyAsync(xbuf, x_in, (size_t)NT * DIMC * sizeof(float),
                 hipMemcpyDeviceToDevice, stream);

  const dim3 blk256(256);
  const long long SHN = (long long)DHEAD * Nseq;     // per-(b,h) stride in kT/vT
  const long long SCT = 16384;                       // per-bucket 128x128

  for (int L = 0; L < DEPTH; L++) {
    // ---- weights -> bf16 transposed ----
    castT_kernel<<<dim3(HD/32,  DIMC/32), dim3(32,8), 0, stream>>>(Wq + (size_t)L*DIMC*HD, wTq, DIMC, HD);
    castT_kernel<<<dim3(HD/32,  DIMC/32), dim3(32,8), 0, stream>>>(Wk + (size_t)L*DIMC*HD, wTk, DIMC, HD);
    castT_kernel<<<dim3(HD/32,  DIMC/32), dim3(32,8), 0, stream>>>(Wv + (size_t)L*DIMC*HD, wTv, DIMC, HD);
    castT_kernel<<<dim3(DIMC/32, HD/32),  dim3(32,8), 0, stream>>>(Wo + (size_t)L*HD*DIMC, wTo, HD, DIMC);
    castT_kernel<<<dim3(DFF/32, DIMC/32), dim3(32,8), 0, stream>>>(W1 + (size_t)L*DIMC*DFF, wT1, DIMC, DFF);
    castT_kernel<<<dim3(DIMC/32, DFF/32), dim3(32,8), 0, stream>>>(W2 + (size_t)L*DFF*DIMC, wT2, DFF, DIMC);

    // ---- LN1 + QKV projections (epilogues split/transpose per operand) ----
    layernorm_kernel<<<NT, blk256, 0, stream>>>(xbuf, ln1g + L*DIMC, ln1b + L*DIMC, hbuf);
    wmma_gemm<128,128,2,4,M_Q>    <<<dim3(HD/128, NT/128, 1), blk256, 0, stream>>>(
        hbuf, DIMC, 0, 0, wTq, DIMC, 0, 0, 1, DIMC, nullptr, qbuf, nullptr, nullptr, nullptr);
    wmma_gemm<128,128,2,4,M_KEXPT><<<dim3(HD/128, NT/128, 1), blk256, 0, stream>>>(
        hbuf, DIMC, 0, 0, wTk, DIMC, 0, 0, 1, DIMC, nullptr, kT, nullptr, nullptr, nullptr);
    wmma_gemm<128,128,2,4,M_VT>   <<<dim3(HD/128, NT/128, 1), blk256, 0, stream>>>(
        hbuf, DIMC, 0, 0, wTv, DIMC, 0, 0, 1, DIMC, nullptr, vT, nullptr, nullptr, nullptr);
    softmax_q_kernel<<<BH * Nseq / 8, blk256, 0, stream>>>(qbuf);

    // ---- attention phase 1: S^T[b,h,u] = v^T(bucket) x k(bucket) ----
    wmma_gemm<128,128,2,4,M_ST><<<dim3(1, 1, BH * U), blk256, 0, stream>>>(
        vT, Nseq, SHN, BUCKET, kT, Nseq, SHN, BUCKET, U, BUCKET,
        STb, nullptr, nullptr, nullptr, nullptr);
    ksum_kernel      <<<BH * U * DHEAD / 256, blk256, 0, stream>>>(kT, ksb);
    cumsum_ctx_kernel<<<BH * 64,              blk256, 0, stream>>>(STb, ctxb);
    cumsum_ks_kernel <<<BH * DHEAD / 256,     blk256, 0, stream>>>(ksb, kcum);
    dinv_kernel      <<<BH * Nseq / 256,      blk256, 0, stream>>>(qbuf, kcum, dvb);

    // ---- attention phase 3: out = (q x ctx) * D^-1, scatter to [b,n,h*e] ----
    wmma_gemm<64,128,2,4,M_AOUT><<<dim3(1, 1, BH * U), blk256, 0, stream>>>(
        qbuf, DHEAD, (long long)Nseq * DHEAD, (long long)BUCKET * DHEAD,
        ctxb, DHEAD, (long long)U * SCT, SCT, U, DHEAD,
        nullptr, abuf, nullptr, nullptr, dvb);

    // ---- output projection + residual ----
    wmma_gemm<128,128,2,4,M_BIASRES><<<dim3(DIMC/128, NT/128, 1), blk256, 0, stream>>>(
        abuf, HD, 0, 0, wTo, HD, 0, 0, 1, HD,
        xbuf, nullptr, bo + L*DIMC, xbuf, nullptr);

    // ---- FFN ----
    layernorm_kernel<<<NT, blk256, 0, stream>>>(xbuf, ln2g + L*DIMC, ln2b + L*DIMC, hbuf);
    wmma_gemm<128,128,2,4,M_GELU><<<dim3(DFF/128, NT/128, 1), blk256, 0, stream>>>(
        hbuf, DIMC, 0, 0, wT1, DIMC, 0, 0, 1, DIMC,
        nullptr, y1, b1 + L*DFF, nullptr, nullptr);
    wmma_gemm<128,128,2,4,M_BIASRES><<<dim3(DIMC/128, NT/128, 1), blk256, 0, stream>>>(
        y1, DFF, 0, 0, wT2, DFF, 0, 0, 1, DFF,
        xbuf, nullptr, b2 + L*DIMC, xbuf, nullptr);
  }
}
